// EncoderRNN_49435073577830
// MI455X (gfx1250) — matrix operand order repaired and checked by reading, not compiled, not run on hardware
//
#include <hip/hip_runtime.h>
#include <hip/hip_bf16.h>

// ---------------------------------------------------------------------------
// Problem constants (from reference): B=128, L=512, E=512, H=512, C=16
// ---------------------------------------------------------------------------
#define BB 128
#define LL 512
#define EE 512
#define HH 512
#define CC 16
#define NEG_INF (-1000000000.0f)

typedef __attribute__((ext_vector_type(16))) __bf16 v16bf;
typedef __attribute__((ext_vector_type(8)))  __bf16 v8bf;
typedef __attribute__((ext_vector_type(8)))  float  v8f;

union Frag { v16bf v; v8bf h[2]; };

__device__ __forceinline__ v8f vzero() {
  v8f z = {0.f,0.f,0.f,0.f,0.f,0.f,0.f,0.f};
  return z;
}
__device__ __forceinline__ float fsigm(float x)  { return 1.0f / (1.0f + __expf(-x)); }
__device__ __forceinline__ float ftanh(float x)  { return 1.0f - 2.0f / (__expf(2.0f * x) + 1.0f); }

// Opaque zero offset (SGPR). Added to a pointer it makes the address
// loop-variant (defeats LICM hoist+spill of loop-invariant global loads)
// WITHOUT breaking address-space inference: the pointer chain stays a GEP on
// the kernarg-derived base, so loads still lower to global_load_* (not flat).
__device__ __forceinline__ size_t opaque_off() {
  unsigned long long q = 0;
  asm volatile("" : "+s"(q));
  return (size_t)q;
}

// ---------------------------------------------------------------------------
// Workspace layout (bytes). Total ~209 MB.
// ---------------------------------------------------------------------------
static constexpr size_t MBy     = 1024ull * 1024ull;
static constexpr size_t OFF_WPK  = 0;            // 8 gate mats packed bf16: 8*32*32*512*2 = 8 MB
static constexpr size_t OFF_AWPK = 8  * MBy;     // aW packed: 32*64*512*2 = 2 MB
static constexpr size_t OFF_CWPK = 10 * MBy;     // cW packed: 32*512*2 = 32 KB
static constexpr size_t OFF_XBF  = 12 * MBy;     // x bf16: 128*512*512*2 = 64 MB
static constexpr size_t OFF_HID  = 76 * MBy;     // hiddens bf16: 128*512*1024*2 = 128 MB
static constexpr size_t OFF_HB   = 204 * MBy;    // h double-buffer: 2dir*2buf*128*512*2 = 512 KB
static constexpr size_t OFF_CB   = 205 * MBy;    // c state fp32: 2*128*512*4 = 512 KB
static constexpr size_t OFF_AL   = 206 * MBy;    // alpha fp32: 128*512*4
static constexpr size_t OFF_SE   = 207 * MBy;    // sent bf16: 128*1024*2
static constexpr size_t OFF_CTR  = 208 * MBy;    // step barrier counters: 2*512*u32

// ---------------------------------------------------------------------------
// K0: zero recurrent state + barrier counters (ws is poisoned by harness)
// ---------------------------------------------------------------------------
__global__ void init_state(__bf16* hbuf, float* cbuf, unsigned* ctr) {
  int i = blockIdx.x * blockDim.x + threadIdx.x;           // 262144 threads
  if (i < 2 * 2 * BB * HH) hbuf[i] = (__bf16)0.0f;         // 2 dirs x 2 bufs
  if (i < 2 * BB * HH)     cbuf[i] = 0.0f;
  if (i < 2 * LL)          ctr[i]  = 0u;
}

// ---------------------------------------------------------------------------
// K1: fp32 -> bf16 conversion of x (one-shot, amortized over 512 steps)
// ---------------------------------------------------------------------------
__global__ void convert_x(const float* __restrict__ x, __bf16* __restrict__ xb, int n) {
  int i = blockIdx.x * blockDim.x + threadIdx.x;
  if (i < n) xb[i] = (__bf16)x[i];
}

// ---------------------------------------------------------------------------
// K2: pack a KxN fp32 matrix into WMMA bf16 B-fragment tile order.
// Tile (kt, nt) is 32x16; within a tile, lane L owns column N = nt*16 + (L&15)
// and K values kt*32 + (L<16 ? 0..15 : 16..31), stored lane-major so each lane
// fetches its 16 bf16 with two contiguous global_load_b128.
// ---------------------------------------------------------------------------
__global__ void pack_b_tiles(const float* __restrict__ src, __bf16* __restrict__ dst,
                             int ktiles, int ntiles, int N) {
  int idx = blockIdx.x * blockDim.x + threadIdx.x;
  int total = ktiles * ntiles * 512;
  if (idx >= total) return;
  int within = idx & 511;
  int tile   = idx >> 9;
  int lane   = within >> 4;
  int i      = within & 15;
  int kt     = tile / ntiles;
  int nt     = tile - kt * ntiles;
  int k = kt * 32 + ((lane < 16) ? 0 : 16) + i;
  int n = nt * 16 + (lane & 15);
  dst[idx] = (__bf16)src[(size_t)k * N + n];
}

// ---------------------------------------------------------------------------
// K3: persistent bidirectional LSTM.
// 64 WGs x 256 thr: blocks [0,32) forward, [32,64) backward.
// Each wave owns one (m_tile, h_tile): 4 gate accumulators (16x16 f32 each),
// K loop = 32 x wmma_f32_16x16x32_bf16 per gate. Grid-wide step barrier via
// per-step atomic counters; h state double-buffered. Weight base gets a fresh
// opaque offset per step so B tiles re-stream from L2 (no scratch spills).
// ---------------------------------------------------------------------------
__global__ void __launch_bounds__(256, 1) lstm_persistent(
    const __bf16* __restrict__ xbf, const __bf16* __restrict__ wpk,
    const float* fbf, const float* fbi, const float* fbo, const float* fbg,
    const float* bbf, const float* bbi, const float* bbo, const float* bbg,
    __bf16* hbuf, float* cbuf, __bf16* hid, unsigned* ctr) {
  const int dir    = blockIdx.x >> 5;
  const int wave   = threadIdx.x >> 5;
  const int lane   = threadIdx.x & 31;
  const int waveId = ((blockIdx.x & 31) << 3) + wave;  // 0..255
  const int mtile  = waveId >> 5;                      // 0..7   (batch tiles)
  const int ntile  = waveId & 31;                      // 0..31  (hidden tiles)
  const int mrow   = mtile * 16 + (lane & 15);         // A-fragment batch row
  const int ncol   = ntile * 16 + (lane & 15);         // output hidden column
  const int koffA  = (lane < 16) ? 0 : 16;

  float bias[4];
  if (dir == 0) { bias[0]=fbf[ncol]; bias[1]=fbi[ncol]; bias[2]=fbo[ncol]; bias[3]=fbg[ncol]; }
  else          { bias[0]=bbf[ncol]; bias[1]=bbi[ncol]; bias[2]=bbo[ncol]; bias[3]=bbg[ncol]; }

  __bf16*       hb    = hbuf + (size_t)dir * 2 * BB * HH;
  float*        cp    = cbuf + (size_t)dir * BB * HH;
  const __bf16* wbase = wpk  + (size_t)dir * 4 * 32 * 32 * 512;
  unsigned*     ctrd  = ctr  + dir * LL;

  for (int tt = 0; tt < LL; ++tt) {
    const int t = dir ? (LL - 1 - tt) : tt;
    const __bf16* hread  = hb + (size_t)(tt & 1) * BB * HH;
    __bf16*       hwrite = hb + (size_t)((tt + 1) & 1) * BB * HH;

    // Fresh opaque offset each step: weight loads cannot be hoisted/spilled,
    // yet remain global-address-space loads.
    const __bf16* wptr = wbase + opaque_off();

    v8f acc[4];
    #pragma unroll
    for (int g = 0; g < 4; ++g) acc[g] = vzero();

    const __bf16* xrow = xbf + ((size_t)mrow * LL + t) * EE + koffA;
    const __bf16* hrow = hread + (size_t)mrow * HH + koffA;

    for (int kt = 0; kt < 32; ++kt) {
      Frag a;
      const __bf16* ar = (kt < 16) ? (xrow + kt * 32) : (hrow + (kt - 16) * 32);
      a.h[0] = *(const v8bf*)(ar);
      a.h[1] = *(const v8bf*)(ar + 8);
      #pragma unroll
      for (int g = 0; g < 4; ++g) {
        const __bf16* br = wptr + ((size_t)((g * 32 + kt) * 32 + ntile)) * 512 + lane * 16;
        Frag bfr;
        bfr.h[0] = *(const v8bf*)(br);
        bfr.h[1] = *(const v8bf*)(br + 8);
        acc[g] = __builtin_amdgcn_wmma_f32_16x16x32_bf16(
            false, a.v, false, bfr.v, (short)0, acc[g], false, false);
      }
    }

    // Fused LSTM cell epilogue. C-layout: VGPR r, lanes 0-15 -> M = mtile*16+r,
    // lanes 16-31 -> M = mtile*16+8+r; N = ncol.
    #pragma unroll
    for (int r = 0; r < 8; ++r) {
      const int m = mtile * 16 + r + ((lane < 16) ? 0 : 8);
      const size_t idx = (size_t)m * HH + ncol;
      const float f = fsigm(acc[0][r] + bias[0]);
      const float i = fsigm(acc[1][r] + bias[1]);
      const float o = fsigm(acc[2][r] + bias[2]);
      const float g = ftanh(acc[3][r] + bias[3]);
      const float cnew = f * cp[idx] + i * g;   // c owned by this wave only
      cp[idx] = cnew;
      const float hv = o * ftanh(cnew);
      hwrite[idx] = (__bf16)hv;
      hid[((size_t)m * LL + t) * (2 * HH) + dir * HH + ncol] = (__bf16)hv;
    }

    // Device-scope step barrier for this direction (32 WGs).
    __threadfence();
    __syncthreads();
    if (threadIdx.x == 0) {
      atomicAdd(&ctrd[tt], 1u);
      while (__hip_atomic_load(&ctrd[tt], __ATOMIC_RELAXED, __HIP_MEMORY_SCOPE_AGENT) < 32u) {
        __builtin_amdgcn_s_sleep(1);
      }
    }
    __syncthreads();
  }
}

// ---------------------------------------------------------------------------
// K4: alpha[b,l] = sum_n tanh((hiddens @ aW)[b,l,n] + ab[n]) * av[n]
// WMMA GEMM (65536x1024 @ 1024x1024) fused with tanh/av reduction so the
// 256 MB attn_f intermediate never hits memory. One 16-row M-tile per wave.
// A-fragment base gets a fresh opaque offset per nt so A loads re-issue
// (L0/L2 hits) instead of being hoisted across the nt loop and spilled.
// ---------------------------------------------------------------------------
__global__ void __launch_bounds__(256, 1) attn_scores(
    const __bf16* __restrict__ hid, const __bf16* __restrict__ awpk,
    const float* __restrict__ ab, const float* __restrict__ av,
    float* __restrict__ alpha) {
  const int wave = threadIdx.x >> 5;
  const int lane = threadIdx.x & 31;
  const int mt   = blockIdx.x * 8 + wave;            // 0..4095
  const int mrow = mt * 16 + (lane & 15);            // row over (b,l)
  const size_t arowoff = (size_t)mrow * (2 * HH) + ((lane < 16) ? 0 : 16);

  float rs[8];
  #pragma unroll
  for (int r = 0; r < 8; ++r) rs[r] = 0.0f;

  for (int nt = 0; nt < 64; ++nt) {
    const __bf16* abase = hid + opaque_off() + arowoff;
    v8f acc = vzero();
    for (int kt = 0; kt < 32; ++kt) {
      Frag a;
      const __bf16* ar = abase + kt * 32;
      a.h[0] = *(const v8bf*)(ar);
      a.h[1] = *(const v8bf*)(ar + 8);
      Frag bfr;
      const __bf16* br = awpk + ((size_t)(kt * 64 + nt)) * 512 + lane * 16;
      bfr.h[0] = *(const v8bf*)(br);
      bfr.h[1] = *(const v8bf*)(br + 8);
      acc = __builtin_amdgcn_wmma_f32_16x16x32_bf16(
          false, a.v, false, bfr.v, (short)0, acc, false, false);
    }
    const int n  = nt * 16 + (lane & 15);
    const float abn = ab[n], avn = av[n];
    #pragma unroll
    for (int r = 0; r < 8; ++r) rs[r] += ftanh(acc[r] + abn) * avn;
  }
  // Reduce each row across its 16-lane half (xor masks stay inside halves).
  #pragma unroll
  for (int r = 0; r < 8; ++r) {
    float s = rs[r];
    for (int msk = 1; msk < 16; msk <<= 1) s += __shfl_xor(s, msk, 32);
    if ((lane & 15) == 0) {
      const int m = mt * 16 + r + ((lane < 16) ? 0 : 8);
      alpha[m] = s;
    }
  }
}

// ---------------------------------------------------------------------------
// K5: per-batch masked softmax over L + attention-weighted sum of hiddens.
// One WG per batch row; ~1 MB of bf16 hiddens streamed per WG (HBM-bound).
// ---------------------------------------------------------------------------
__global__ void __launch_bounds__(256) attn_softmax_sent(
    const float* __restrict__ alpha, const int* __restrict__ mask,
    const __bf16* __restrict__ hid, __bf16* __restrict__ sent) {
  __shared__ float w[LL];
  __shared__ float red[256];
  const int b = blockIdx.x, tid = threadIdx.x;

  float a0 = alpha[b * LL + tid];
  float a1 = alpha[b * LL + tid + 256];
  if (mask[b * LL + tid] == 0)       a0 = NEG_INF;
  if (mask[b * LL + tid + 256] == 0) a1 = NEG_INF;

  red[tid] = fmaxf(a0, a1);
  __syncthreads();
  for (int s = 128; s > 0; s >>= 1) { if (tid < s) red[tid] = fmaxf(red[tid], red[tid + s]); __syncthreads(); }
  const float mx = red[0];
  __syncthreads();

  const float e0 = __expf(a0 - mx), e1 = __expf(a1 - mx);
  w[tid] = e0; w[tid + 256] = e1;
  red[tid] = e0 + e1;
  __syncthreads();
  for (int s = 128; s > 0; s >>= 1) { if (tid < s) red[tid] += red[tid + s]; __syncthreads(); }
  const float inv = 1.0f / red[0];
  __syncthreads();

  float acc[4] = {0.f, 0.f, 0.f, 0.f};
  for (int l = 0; l < LL; ++l) {
    const float wl = w[l] * inv;
    const __bf16* hrow = hid + ((size_t)b * LL + l) * (2 * HH);
    acc[0] += wl * (float)hrow[tid];
    acc[1] += wl * (float)hrow[tid + 256];
    acc[2] += wl * (float)hrow[tid + 512];
    acc[3] += wl * (float)hrow[tid + 768];
  }
  sent[b * (2 * HH) + tid]       = (__bf16)acc[0];
  sent[b * (2 * HH) + tid + 256] = (__bf16)acc[1];
  sent[b * (2 * HH) + tid + 512] = (__bf16)acc[2];
  sent[b * (2 * HH) + tid + 768] = (__bf16)acc[3];
}

// ---------------------------------------------------------------------------
// K6: logits = sent @ cW + cb, then log_softmax over C=16 via lane shuffles.
// One WG, 8 waves = 8 M-tiles; single N-tile.
// ---------------------------------------------------------------------------
__global__ void __launch_bounds__(256, 1) logits_lsm(
    const __bf16* __restrict__ sent, const __bf16* __restrict__ cwpk,
    const float* __restrict__ cb, float* __restrict__ out) {
  const int wave = threadIdx.x >> 5;
  const int lane = threadIdx.x & 31;
  const int mt   = wave;
  const int mrow = mt * 16 + (lane & 15);
  const int koffA = (lane < 16) ? 0 : 16;

  v8f acc = vzero();
  for (int kt = 0; kt < 32; ++kt) {
    Frag a;
    const __bf16* ar = sent + (size_t)mrow * (2 * HH) + kt * 32 + koffA;
    a.h[0] = *(const v8bf*)(ar);
    a.h[1] = *(const v8bf*)(ar + 8);
    Frag bfr;
    const __bf16* br = cwpk + (size_t)kt * 512 + lane * 16;
    bfr.h[0] = *(const v8bf*)(br);
    bfr.h[1] = *(const v8bf*)(br + 8);
    acc = __builtin_amdgcn_wmma_f32_16x16x32_bf16(
        false, a.v, false, bfr.v, (short)0, acc, false, false);
  }
  const int n = lane & 15;
  const float cbn = cb[n];
  #pragma unroll
  for (int r = 0; r < 8; ++r) {
    const float v = acc[r] + cbn;
    float mxv = v;
    for (int msk = 1; msk < 16; msk <<= 1) mxv = fmaxf(mxv, __shfl_xor(mxv, msk, 32));
    float s = __expf(v - mxv);
    for (int msk = 1; msk < 16; msk <<= 1) s += __shfl_xor(s, msk, 32);
    const int m = mt * 16 + r + ((lane < 16) ? 0 : 8);
    out[m * CC + n] = v - (mxv + __logf(s));
  }
}

// ---------------------------------------------------------------------------
// Host: deterministic launch sequence, all on `stream` (graph-capture safe).
// ---------------------------------------------------------------------------
extern "C" void kernel_launch(void* const* d_in, const int* in_sizes, int n_in,
                              void* d_out, int out_size, void* d_ws, size_t ws_size,
                              hipStream_t stream) {
  const float* x    = (const float*)d_in[0];
  const int*   mask = (const int*)d_in[1];
  // gate weights/biases: fWf=2 fbf=3 fWi=4 fbi=5 fWo=6 fbo=7 fWg=8 fbg=9, b*: 10..17
  const float* aW = (const float*)d_in[18];
  const float* ab = (const float*)d_in[19];
  const float* av = (const float*)d_in[20];
  const float* cW = (const float*)d_in[21];
  const float* cb = (const float*)d_in[22];

  char* ws = (char*)d_ws;
  __bf16*   wpk   = (__bf16*)(ws + OFF_WPK);
  __bf16*   awpk  = (__bf16*)(ws + OFF_AWPK);
  __bf16*   cwpk  = (__bf16*)(ws + OFF_CWPK);
  __bf16*   xbf   = (__bf16*)(ws + OFF_XBF);
  __bf16*   hid   = (__bf16*)(ws + OFF_HID);
  __bf16*   hbuf  = (__bf16*)(ws + OFF_HB);
  float*    cbuf  = (float*)(ws + OFF_CB);
  float*    alpha = (float*)(ws + OFF_AL);
  __bf16*   sent  = (__bf16*)(ws + OFF_SE);
  unsigned* ctr   = (unsigned*)(ws + OFF_CTR);

  // K0: zero state + counters (ws poisoned by harness; must re-init per call)
  init_state<<<1024, 256, 0, stream>>>(hbuf, cbuf, ctr);

  // K1: x -> bf16
  const int nX = BB * LL * EE;
  convert_x<<<nX / 256, 256, 0, stream>>>(x, xbf, nX);

  // K2: pack all B matrices into WMMA fragment order
  for (int dir = 0; dir < 2; ++dir)
    for (int g = 0; g < 4; ++g) {
      const float* src = (const float*)d_in[2 + dir * 8 + g * 2];
      __bf16* dst = wpk + (size_t)(dir * 4 + g) * 32 * 32 * 512;
      pack_b_tiles<<<(32 * 32 * 512) / 256, 256, 0, stream>>>(src, dst, 32, 32, HH);
    }
  pack_b_tiles<<<(32 * 64 * 512) / 256, 256, 0, stream>>>(aW, awpk, 32, 64, 2 * HH);
  pack_b_tiles<<<(32 * 1 * 512) / 256, 256, 0, stream>>>(cW, cwpk, 32, 1, CC);

  // K3: persistent bidirectional LSTM (32 WGs per direction)
  lstm_persistent<<<64, 256, 0, stream>>>(
      xbf, wpk,
      (const float*)d_in[3],  (const float*)d_in[5],  (const float*)d_in[7],  (const float*)d_in[9],
      (const float*)d_in[11], (const float*)d_in[13], (const float*)d_in[15], (const float*)d_in[17],
      hbuf, cbuf, hid, ctr);

  // K4: attention scores (fused GEMM + tanh + av dot)
  attn_scores<<<512, 256, 0, stream>>>(hid, awpk, ab, av, alpha);

  // K5: masked softmax + weighted sum -> sent
  attn_softmax_sent<<<BB, 256, 0, stream>>>(alpha, mask, hid, sent);

  // K6: classifier + log_softmax -> d_out
  logits_lsm<<<1, 256, 0, stream>>>(sent, cwpk, cb, (float*)d_out);
}
